// ModularLinear_21930103013543
// MI455X (gfx1250) — matrix operand order repaired and checked by reading, not compiled, not run on hardware
//
#include <hip/hip_runtime.h>

typedef __bf16 bf16_t;
typedef __attribute__((ext_vector_type(16))) __bf16 v16bf;
typedef __attribute__((ext_vector_type(8)))  __bf16 v8bf;
typedef __attribute__((ext_vector_type(8)))  float  v8f;
typedef __attribute__((ext_vector_type(4)))  unsigned int v4u;
typedef __attribute__((ext_vector_type(8)))  int v8i;
typedef __attribute__((ext_vector_type(4)))  int v4i;

#define T_DIM    2048
#define B_DIM    8
#define IN_DIM   1024
#define OUT_DIM  1024
#define KSEL     2
#define TILE_K   32
#define LDS_STRIDE 40   // 32 + 8 pad; 80B row stride keeps 16B alignment, breaks bank conflicts

#if defined(__gfx1250__) && __has_builtin(__builtin_amdgcn_tensor_load_to_lds) && \
    __has_builtin(__builtin_amdgcn_s_wait_tensorcnt)
#define USE_TDM 1
#else
#define USE_TDM 0
#endif

static __device__ __forceinline__ v16bf cat16(v8bf a, v8bf b) {
    return __builtin_shufflevector(a, b, 0,1,2,3,4,5,6,7,8,9,10,11,12,13,14,15);
}

// split f32 into bf16 hi + bf16 residual lo, store both to LDS
static __device__ __forceinline__ void split_store(bf16_t* hi, bf16_t* lo, int off, float v) {
    bf16_t h = (bf16_t)v;
    hi[off] = h;
    lo[off] = (bf16_t)(v - (float)h);
}

#if USE_TDM
// TDM: load one 128-row x 32-col fp32 tile (row stride in elements) into LDS,
// rows stored contiguously (32 floats / 128B per row). D# per ISA 08_async_tensor §8.
// This toolchain exposes the 6-arg builtin: (v4u g0, v8i g1, v4i g2, v4i g3, v8i g4, i32 cpol);
// groups 2/3/4 are unused for a plain 2D tile -> zeros.
static __device__ __forceinline__ void tdm_load_tile_128x32_f32(
    unsigned lds_byte_addr, const float* gptr, unsigned row_stride_elems)
{
    unsigned long long ga = (unsigned long long)(uintptr_t)gptr;
    v4u g0;
    g0.x = 1u;                                         // count=1, user mode
    g0.y = lds_byte_addr;                              // LDS byte address
    g0.z = (unsigned)(ga & 0xFFFFFFFFu);               // global_addr[31:0]
    g0.w = (unsigned)((ga >> 32) & 0x01FFFFFFu)        // global_addr[56:32]
         | (2u << 30);                                 // type = 2 ("image")
    v8i g1;
    g1[0] = (int)(2u << 16);                           // data_size = 2 (4 bytes); no multicast/pad/iterate
    g1[1] = (int)(0xFFFFu << 16);                      // tensor_dim0[15:0]  (dim0 = 0x7FFFFFFF: never OOB)
    g1[2] = (int)(0x7FFFu | (0xFFFFu << 16));          // tensor_dim0[31:16] | tensor_dim1[15:0]
    g1[3] = (int)(0x7FFFu | (32u << 16));              // tensor_dim1[31:16] | tile_dim0 = 32
    g1[4] = (int)128;                                  // tile_dim1 = 128, tile_dim2 = 0
    g1[5] = (int)row_stride_elems;                     // tensor_dim0_stride[31:0]
    g1[6] = 0;                                         // stride0[47:32]=0 | stride1[15:0]=0
    g1[7] = 0;                                         // stride1[47:16]=0
    v4i z4 = {0, 0, 0, 0};                             // 2D tensor: groups 2/3 unused
    v8i z8 = {0, 0, 0, 0, 0, 0, 0, 0};                 // extra group (6-arg form): unused
    __builtin_amdgcn_tensor_load_to_lds(g0, g1, z4, z4, z8, 0);
}
#endif

__global__ __launch_bounds__(256)
void modlin_wmma_bf16x3(const float* __restrict__ x,
                        const int*   __restrict__ sel,
                        const float* __restrict__ w,
                        float*       __restrict__ out)
{
    __shared__ bf16_t sXhi[128 * LDS_STRIDE];
    __shared__ bf16_t sXlo[128 * LDS_STRIDE];
    __shared__ bf16_t sWhi[128 * LDS_STRIDE];
    __shared__ bf16_t sWlo[128 * LDS_STRIDE];
#if USE_TDM
    __shared__ float stX[2][128 * TILE_K];             // double-buffered fp32 staging (TDM dest)
    __shared__ float stW[2][128 * TILE_K];
#endif

    const int tid  = threadIdx.x;
    const int lane = tid & 31;
    const int wid  = tid >> 5;
    const int wm   = wid & 3;      // 4 waves tile M (time)
    const int wn   = wid >> 2;     // 2 waves tile N (out)
    const int lr   = lane & 15;    // row/col within 16x16 tile
    const int lh   = lane >> 4;    // lane half selects K sub-range

    const int slot = blockIdx.z;   // 0..15 = (b, k-slot)
    const int b    = slot >> 1;
    const int ks   = slot & 1;
    const int e    = sel[b * KSEL + ks];

    const int t0 = blockIdx.y * 128;
    const int o0 = blockIdx.x * 128;

    const float* xBase = x + (size_t)t0 * (B_DIM * IN_DIM) + (size_t)b * IN_DIM;
    const float* wBase = w + (size_t)e * ((size_t)OUT_DIM * IN_DIM) + (size_t)o0 * IN_DIM;

    v8f acc[2][4];
    #pragma unroll
    for (int mt = 0; mt < 2; ++mt)
        #pragma unroll
        for (int nt = 0; nt < 4; ++nt) {
            v8f z = {};
            acc[mt][nt] = z;
        }

    const int NK = IN_DIM / TILE_K;  // 32 k-steps

#if USE_TDM
    // LDS flat aperture keeps the offset in addr[31:0] -> truncation yields D#.lds_addr
    const unsigned aX[2] = { (unsigned)(uintptr_t)&stX[0][0], (unsigned)(uintptr_t)&stX[1][0] };
    const unsigned aW[2] = { (unsigned)(uintptr_t)&stW[0][0], (unsigned)(uintptr_t)&stW[1][0] };
    if (wid == 0) {  // prologue: fill buffer 0 (TDM ignores EXEC; one issuing wave)
        tdm_load_tile_128x32_f32(aX[0], xBase, B_DIM * IN_DIM);
        tdm_load_tile_128x32_f32(aW[0], wBase, IN_DIM);
    }
#endif

    for (int ik = 0; ik < NK; ++ik) {
        const int kc = ik * TILE_K;
#if USE_TDM
        const int ib = ik & 1;
        if (wid == 0) {
            if (ik + 1 < NK) {   // prefetch next panel pair into the other buffer
                tdm_load_tile_128x32_f32(aX[ib ^ 1], xBase + kc + TILE_K, B_DIM * IN_DIM);
                tdm_load_tile_128x32_f32(aW[ib ^ 1], wBase + kc + TILE_K, IN_DIM);
                __builtin_amdgcn_s_wait_tensorcnt((short)2);  // current pair done, next 2 in flight
            } else {
                __builtin_amdgcn_s_wait_tensorcnt((short)0);
            }
        }
        __syncthreads();  // (a) staging[ib] visible to all waves; prev frag reads done -> panels free

        // ---- convert fp32 staging -> (bf16 hi, bf16 lo) panels; 4 float4 per thread per panel ----
        #pragma unroll
        for (int i = 0; i < 4; ++i) {
            const int idx = tid + i * 256;          // 0..1023
            const int r   = idx >> 3;               // row 0..127
            const int c   = (idx & 7) << 2;         // col 0,4,...,28
            const float4 xv = *(const float4*)&stX[ib][r * TILE_K + c];
            const float4 wv = *(const float4*)&stW[ib][r * TILE_K + c];
            const int o = r * LDS_STRIDE + c;
            split_store(sXhi, sXlo, o + 0, xv.x);
            split_store(sXhi, sXlo, o + 1, xv.y);
            split_store(sXhi, sXlo, o + 2, xv.z);
            split_store(sXhi, sXlo, o + 3, xv.w);
            split_store(sWhi, sWlo, o + 0, wv.x);
            split_store(sWhi, sWlo, o + 1, wv.y);
            split_store(sWhi, sWlo, o + 2, wv.z);
            split_store(sWhi, sWlo, o + 3, wv.w);
        }
        __syncthreads();  // (b) panels ready; staging[ib] reads complete -> safe to overwrite
#else
        // ---- fallback: cooperative global->LDS with f32 -> (bf16 hi, lo) split ----
        #pragma unroll
        for (int i = 0; i < 4; ++i) {
            const int idx = tid + i * 256;
            const int r   = idx >> 3;
            const int c   = (idx & 7) << 2;
            const float4 xv = *(const float4*)(xBase + (size_t)r * (B_DIM * IN_DIM) + kc + c);
            const float4 wv = *(const float4*)(wBase + (size_t)r * IN_DIM + kc + c);
            const int o = r * LDS_STRIDE + c;
            split_store(sXhi, sXlo, o + 0, xv.x);
            split_store(sXhi, sXlo, o + 1, xv.y);
            split_store(sXhi, sXlo, o + 2, xv.z);
            split_store(sXhi, sXlo, o + 3, xv.w);
            split_store(sWhi, sWlo, o + 0, wv.x);
            split_store(sWhi, sWlo, o + 1, wv.y);
            split_store(sWhi, sWlo, o + 2, wv.z);
            split_store(sWhi, sWlo, o + 3, wv.w);
        }
        const int knext = kc + TILE_K;
        if (knext < IN_DIM) {
            const int pr = tid >> 1;
            const int ph = (tid & 1) * 16;
            __builtin_prefetch(xBase + (size_t)pr * (B_DIM * IN_DIM) + knext + ph, 0, 1);
            __builtin_prefetch(wBase + (size_t)pr * IN_DIM + knext + ph, 0, 1);
        }
        __syncthreads();
#endif

        // ---- load fragments from LDS ----
        // A (16x32 bf16): lane half 0 -> K 0..15, half 1 -> K 16..31 (contiguous)
        v16bf aHi[2], aLo[2];
        #pragma unroll
        for (int mt = 0; mt < 2; ++mt) {
            const int row = wm * 32 + mt * 16 + lr;
            const bf16_t* ph = &sXhi[row * LDS_STRIDE];
            const bf16_t* pl = &sXlo[row * LDS_STRIDE];
            aHi[mt] = cat16(*(const v8bf*)(ph + lh * 16), *(const v8bf*)(ph + lh * 16 + 8));
            aLo[mt] = cat16(*(const v8bf*)(pl + lh * 16), *(const v8bf*)(pl + lh * 16 + 8));
        }
        // B (32x16 bf16): lane = column; half 0 -> K 0..15, half 1 -> K 16..31 (contiguous)
        v16bf bHi[4], bLo[4];
        #pragma unroll
        for (int nt = 0; nt < 4; ++nt) {
            const int col = wn * 64 + nt * 16 + lr;
            const bf16_t* ph = &sWhi[col * LDS_STRIDE + lh * 16];
            const bf16_t* pl = &sWlo[col * LDS_STRIDE + lh * 16];
            bHi[nt] = cat16(*(const v8bf*)(ph), *(const v8bf*)(ph + 8));
            bLo[nt] = cat16(*(const v8bf*)(pl), *(const v8bf*)(pl + 8));
        }
#if !USE_TDM
        __syncthreads();   // frag reads done; next iteration may overwrite panels
#endif

        // ---- split-bf16 MMA: D += Ahi*Bhi + Ahi*Blo + Alo*Bhi (~fp32 accuracy) ----
        #pragma unroll
        for (int mt = 0; mt < 2; ++mt) {
            #pragma unroll
            for (int nt = 0; nt < 4; ++nt) {
                acc[mt][nt] = __builtin_amdgcn_wmma_f32_16x16x32_bf16(
                    false, aHi[mt], false, bHi[nt], (short)0, acc[mt][nt], false, false);
                acc[mt][nt] = __builtin_amdgcn_wmma_f32_16x16x32_bf16(
                    false, aHi[mt], false, bLo[nt], (short)0, acc[mt][nt], false, false);
                acc[mt][nt] = __builtin_amdgcn_wmma_f32_16x16x32_bf16(
                    false, aLo[mt], false, bHi[nt], (short)0, acc[mt][nt], false, false);
            }
        }
    }

    // ---- writeback: C layout = 8 VGPRs, VGPR r -> M = r + 8*lanehalf, lane[3:0] -> N ----
    #pragma unroll
    for (int mt = 0; mt < 2; ++mt) {
        #pragma unroll
        for (int nt = 0; nt < 4; ++nt) {
            #pragma unroll
            for (int r = 0; r < 8; ++r) {
                const int m = t0 + wm * 32 + mt * 16 + r + lh * 8;    // global t
                const int n = o0 + wn * 64 + nt * 16 + lr;            // global out col
                out[((size_t)m * B_DIM + b) * (KSEL * OUT_DIM) + (size_t)ks * OUT_DIM + n] =
                    acc[mt][nt][r];
            }
        }
    }
}

extern "C" void kernel_launch(void* const* d_in, const int* in_sizes, int n_in,
                              void* d_out, int out_size, void* d_ws, size_t ws_size,
                              hipStream_t stream) {
    (void)in_sizes; (void)n_in; (void)out_size; (void)d_ws; (void)ws_size;
    const float* x   = (const float*)d_in[0];
    const int*   sel = (const int*)d_in[1];
    const float* w   = (const float*)d_in[2];
    float*       out = (float*)d_out;

    dim3 grid(OUT_DIM / 128, T_DIM / 128, B_DIM * KSEL);  // 8 x 16 x 16 = 2048 blocks
    dim3 block(256);                                       // 8 wave32
    hipLaunchKernelGGL(modlin_wmma_bf16x3, grid, block, 0, stream, x, sel, w, out);
}
